// CouplingChannelGAT_64355789963665
// MI455X (gfx1250) — matrix operand order repaired and checked by reading, not compiled, not run on hardware
//
#include <hip/hip_runtime.h>
#include <hip/hip_bf16.h>
#include <math.h>
#include <cstring>

// ---------------- problem constants (match reference) ----------------
#define N_NODES 202752
#define E_EDGES 1000000
#define F_IN    64
#define H       128
#define HEADS   4
#define B_GRAPHS 2048
#define NUM_CH  6
#define NSEG    (B_GRAPHS * NUM_CH)   // 12288
#define FF_DIM  512
#define RO_HID  256
#define NEGC    (-1e9f)
#define LN_EPS  1e-5f

typedef _Float16 h4   __attribute__((ext_vector_type(4)));
typedef _Float16 h8   __attribute__((ext_vector_type(8)));
typedef _Float16 v16h __attribute__((ext_vector_type(16)));
typedef float    v8f  __attribute__((ext_vector_type(8)));

// ---------------- helpers ----------------
__device__ __forceinline__ float warp_sum(float v) {
#pragma unroll
  for (int o = 16; o > 0; o >>= 1) v += __shfl_xor(v, o, 32);
  return v;
}
__device__ __forceinline__ float eluf(float x) { return x > 0.f ? x : expm1f(x); }

// monotonic float<->uint encoding so atomicMax(uint) == float max
__device__ __forceinline__ unsigned fenc(float x) {
  unsigned u = __float_as_uint(x);
  return (u & 0x80000000u) ? ~u : (u | 0x80000000u);
}
__device__ __forceinline__ float fdec(unsigned e) {
  return __uint_as_float((e & 0x80000000u) ? (e ^ 0x80000000u) : ~e);
}

// ---------------- generic WMMA GEMM: C = A(MxK) @ W(KxN) [+ bias] ----------------
// f32 in global, f16 staged in LDS, f32 accumulate via v_wmma_f32_16x16x32_f16.
// Block = 128 threads = 4 waves; block tile 64(M) x 64(N); K chunked by 64.
// Software-pipelined: next K-chunk is fetched as float4 vector loads into
// registers while the current chunk is consumed from LDS by the WMMAs;
// chunk-after-next is touched via global_prefetch_b8.
// Requires K % 64 == 0 and N % 64 == 0 (true for all layers here).
__global__ __launch_bounds__(128) void k_gemm_f16(
    const float* __restrict__ A, const float* __restrict__ W,
    const float* __restrict__ bias, float* __restrict__ C,
    int M, int K, int Nc) {
  __shared__ _Float16 sA[64][72];   // +8 halves pad: kills LDS bank conflicts
  __shared__ _Float16 sW[64][72];   // stored transposed: sW[n][k]

  const int row0  = blockIdx.x * 64;
  const int col0  = blockIdx.y * 64;
  const int tid   = threadIdx.x;
  const int lane  = tid & 31;
  const int wave  = tid >> 5;
  const int halfid = lane >> 4;     // wave32: lanes 0-15 vs 16-31
  const int l16    = lane & 15;

  // staging decomposition: thread covers rows {(tid>>4) + 8j}, 4-wide column
  // slice at kq = (tid&15)*4  (float4 per row for both A and W)
  const int srow = tid >> 4;        // 0..7
  const int kq   = (tid & 15) * 4;  // 0,4,...,60

  v8f acc[4] = {};                  // 4 n-tiles of 16x16 f32
  float4 aR[8], wR[8];

  auto loadChunk = [&](int k0) {
#pragma unroll
    for (int j = 0; j < 8; ++j) {
      int r = srow + j * 8;
      int gr = row0 + r;
      if (gr < M) aR[j] = *(const float4*)&A[(size_t)gr * K + (k0 + kq)];
      else        aR[j] = make_float4(0.f, 0.f, 0.f, 0.f);
      // W chunk row (k0+r), 4 consecutive columns at col0+kq
      wR[j] = *(const float4*)&W[(size_t)(k0 + r) * Nc + (col0 + kq)];
    }
  };
  auto storeChunk = [&]() {
#pragma unroll
    for (int j = 0; j < 8; ++j) {
      int r = srow + j * 8;
      h4 ap; ap[0] = (_Float16)aR[j].x; ap[1] = (_Float16)aR[j].y;
             ap[2] = (_Float16)aR[j].z; ap[3] = (_Float16)aR[j].w;
      *(h4*)&sA[r][kq] = ap;                       // ds_store_b64, 8B aligned
      sW[kq + 0][r] = (_Float16)wR[j].x;           // transposed scatter
      sW[kq + 1][r] = (_Float16)wR[j].y;
      sW[kq + 2][r] = (_Float16)wR[j].z;
      sW[kq + 3][r] = (_Float16)wR[j].w;
    }
  };

  loadChunk(0);
  const int arow = wave * 16 + l16;   // A: M = lane&15 (ISA 7.12.2)

  for (int k0 = 0; k0 < K; k0 += 64) {
    storeChunk();
    __syncthreads();
    if (k0 + 64 < K) {
      loadChunk(k0 + 64);             // global loads in flight during WMMAs
      if (k0 + 128 < K) {             // speculative prefetch of chunk-after-next
        int gr = row0 + srow;
        __builtin_prefetch(&A[(size_t)((gr < M) ? gr : 0) * K + (k0 + 128 + kq)], 0, 0);
        __builtin_prefetch(&W[(size_t)(k0 + 128 + srow) * Nc + (col0 + kq)], 0, 0);
      }
    }

#pragma unroll
    for (int kk = 0; kk < 64; kk += 32) {
      // A 16x32 f16 fragment: lane half selects K subgroup; two contiguous 8-half chunks
      h8 alo = *(const h8*)&sA[arow][kk + halfid * 8];
      h8 ahi = *(const h8*)&sA[arow][kk + 16 + halfid * 8];
      v16h af;
#pragma unroll
      for (int j = 0; j < 8; ++j) { af[j] = alo[j]; af[j + 8] = ahi[j]; }
      // gather ALL B fragments first so WMMAs don't stall on s_wait_dscnt 0
      v16h bf[4];
#pragma unroll
      for (int t = 0; t < 4; ++t) {
        // B 32x16: lane holds column n = lane&15; k = halfid*8 + e, 16 + halfid*8 + e (matches A)
        const _Float16* bp = &sW[t * 16 + l16][kk + halfid * 8];
        h8 blo = *(const h8*)(bp);
        h8 bhi = *(const h8*)(bp + 16);
#pragma unroll
        for (int j = 0; j < 8; ++j) { bf[t][j] = blo[j]; bf[t][j + 8] = bhi[j]; }
      }
#pragma unroll
      for (int t = 0; t < 4; ++t)
        acc[t] = __builtin_amdgcn_wmma_f32_16x16x32_f16(
            false, af, false, bf[t], (short)0, acc[t], false, false);
    }
    __syncthreads();
  }

  // C/D layout: VGPR r -> row (halfid*8 + r), lane&15 -> column
#pragma unroll
  for (int t = 0; t < 4; ++t) {
    int gcol = col0 + t * 16 + l16;
    float bv = bias ? bias[gcol] : 0.0f;
#pragma unroll
    for (int r = 0; r < 8; ++r) {
      int grow = row0 + wave * 16 + halfid * 8 + r;
      if (grow < M) C[(size_t)grow * Nc + gcol] = acc[t][r] + bv;
    }
  }
}

// ---------------- row-wise LayerNorm (H=128), optional residual-add + ELU ----------------
__global__ __launch_bounds__(256) void k_ln(
    float* __restrict__ X, const float* __restrict__ add,
    const float* __restrict__ g, const float* __restrict__ b,
    int rows, int do_elu) {
  int row = blockIdx.x * 8 + (threadIdx.x >> 5);
  if (row >= rows) return;
  int lane = threadIdx.x & 31;
  size_t base = (size_t)row * H + lane * 4;
  float4 v = *(const float4*)&X[base];
  if (add) {
    float4 a = *(const float4*)&add[base];
    v.x += a.x; v.y += a.y; v.z += a.z; v.w += a.w;
  }
  float mean = warp_sum(v.x + v.y + v.z + v.w) * (1.0f / H);
  float dx = v.x - mean, dy = v.y - mean, dz = v.z - mean, dw = v.w - mean;
  float var = warp_sum(dx * dx + dy * dy + dz * dz + dw * dw) * (1.0f / H);
  float inv = rsqrtf(var + LN_EPS);
  float4 gg = *(const float4*)&g[lane * 4];
  float4 bb = *(const float4*)&b[lane * 4];
  float4 o;
  o.x = dx * inv * gg.x + bb.x; o.y = dy * inv * gg.y + bb.y;
  o.z = dz * inv * gg.z + bb.z; o.w = dw * inv * gg.w + bb.w;
  if (do_elu) { o.x = eluf(o.x); o.y = eluf(o.y); o.z = eluf(o.z); o.w = eluf(o.w); }
  *(float4*)&X[base] = o;
}

// ---------------- GAT per-node attention scores (wave/node, 4 heads of 32) ----------------
__global__ __launch_bounds__(256) void k_gat_scores(
    const float* __restrict__ hW, const float* __restrict__ a_src,
    const float* __restrict__ a_dst, float* __restrict__ as_,
    float* __restrict__ ad_, int n) {
  int node = blockIdx.x * 8 + (threadIdx.x >> 5);
  if (node >= n) return;
  int lane = threadIdx.x & 31;
  int head = lane >> 3;                    // 8 lanes per head
  int off = lane * 4;                      // == head*32 + (lane&7)*4
  float4 hv = *(const float4*)&hW[(size_t)node * H + off];
  float4 ws = *(const float4*)&a_src[off];
  float4 wd = *(const float4*)&a_dst[off];
  float ps = hv.x * ws.x + hv.y * ws.y + hv.z * ws.z + hv.w * ws.w;
  float pd = hv.x * wd.x + hv.y * wd.y + hv.z * wd.z + hv.w * wd.w;
#pragma unroll
  for (int o = 4; o > 0; o >>= 1) { ps += __shfl_xor(ps, o, 32); pd += __shfl_xor(pd, o, 32); }
  if ((lane & 7) == 0) { as_[node * 4 + head] = ps; ad_[node * 4 + head] = pd; }
}

// ---------------- edge passes ----------------
__global__ __launch_bounds__(256) void k_edge_max(
    const int* __restrict__ src, const int* __restrict__ dst, const int* __restrict__ ch,
    const float* __restrict__ as_, const float* __restrict__ ad_,
    unsigned* __restrict__ mx, int E) {
  int e = blockIdx.x * 256 + threadIdx.x;
  if (e >= E) return;
  int s = src[e], d = dst[e];
  if (ch[s] != ch[d]) return;
#pragma unroll
  for (int h = 0; h < HEADS; ++h) {
    float v = as_[s * 4 + h] + ad_[d * 4 + h];
    v = v > 0.f ? v : 0.2f * v;             // leaky_relu(0.2)
    atomicMax(&mx[d * 4 + h], fenc(v));
  }
}

__global__ __launch_bounds__(256) void k_edge_exp(
    const int* __restrict__ src, const int* __restrict__ dst, const int* __restrict__ ch,
    const float* __restrict__ as_, const float* __restrict__ ad_,
    const unsigned* __restrict__ mx, float* __restrict__ exbuf,
    float* __restrict__ den, int E) {
  int e = blockIdx.x * 256 + threadIdx.x;
  if (e >= E) return;
  int s = src[e], d = dst[e];
  bool ok = (ch[s] == ch[d]);
#pragma unroll
  for (int h = 0; h < HEADS; ++h) {
    float ex = 0.f;
    if (ok) {
      float v = as_[s * 4 + h] + ad_[d * 4 + h];
      v = v > 0.f ? v : 0.2f * v;
      ex = expf(v - fdec(mx[d * 4 + h]));
      atomicAdd(&den[d * 4 + h], ex);
    }
    exbuf[(size_t)e * 4 + h] = ex;          // masked edges write 0 (deterministic)
  }
}

__global__ __launch_bounds__(256) void k_edge_agg(
    const int* __restrict__ src, const int* __restrict__ dst, const int* __restrict__ ch,
    const float* __restrict__ hW, const float* __restrict__ exbuf,
    const float* __restrict__ den, float* __restrict__ acc, int E) {
  int e = blockIdx.x * 8 + (threadIdx.x >> 5);   // wave per edge
  if (e >= E) return;
  int lane = threadIdx.x & 31;
  int s = src[e], d = dst[e];
  if (ch[s] != ch[d]) return;
  int h = lane >> 3;                              // features lane*4..+3 all in head lane>>3
  float alpha = exbuf[(size_t)e * 4 + h] / (den[d * 4 + h] + 1e-16f);
  float4 hv = *(const float4*)&hW[(size_t)s * H + lane * 4];
  float* ap = &acc[(size_t)d * H + lane * 4];     // 104MB dest: fully L2-resident
  atomicAdd(ap + 0, alpha * hv.x);
  atomicAdd(ap + 1, alpha * hv.y);
  atomicAdd(ap + 2, alpha * hv.z);
  atomicAdd(ap + 3, alpha * hv.w);
}

__global__ void k_gat_post(float* __restrict__ h, const float* __restrict__ acc,
                           const float* __restrict__ b, size_t total) {
  size_t i = (size_t)blockIdx.x * blockDim.x + threadIdx.x;
  size_t st = (size_t)gridDim.x * blockDim.x;
  for (; i < total; i += st) {
    int col = (int)(i & (H - 1));
    h[i] = eluf(acc[i] + b[col]) + h[i];          // ELU(agg + b) + residual, in-place
  }
}

// ---------------- readout ----------------
__global__ __launch_bounds__(256) void k_ro_score(
    const float* __restrict__ T, const float* __restrict__ w2,
    const float* __restrict__ b2, float* __restrict__ s, int n) {
  int node = blockIdx.x * 8 + (threadIdx.x >> 5);
  if (node >= n) return;
  int lane = threadIdx.x & 31;
  float a = 0.f;
  for (int j = lane; j < H / 2; j += 32) a += tanhf(T[(size_t)node * (H / 2) + j]) * w2[j];
  a = warp_sum(a);
  if (lane == 0) s[node] = a + b2[0];
}

__global__ __launch_bounds__(256) void k_seg_max(
    const float* __restrict__ s, const int* __restrict__ batch, const int* __restrict__ ch,
    unsigned* __restrict__ mxseg, int n) {
  int i = blockIdx.x * 256 + threadIdx.x;
  if (i >= n) return;
  atomicMax(&mxseg[batch[i] * NUM_CH + ch[i]], fenc(s[i]));
}

__global__ __launch_bounds__(256) void k_seg_expsum(
    const float* __restrict__ s, const int* __restrict__ batch, const int* __restrict__ ch,
    const unsigned* __restrict__ mxseg, float* __restrict__ exn,
    float* __restrict__ se, int n) {
  int i = blockIdx.x * 256 + threadIdx.x;
  if (i >= n) return;
  int seg = batch[i] * NUM_CH + ch[i];
  float m = fdec(mxseg[seg]);
  if (!(m > -1e38f)) m = 0.f;                     // empty-segment guard (reference isfinite)
  float ex = expf(s[i] - m);
  exn[i] = ex;
  atomicAdd(&se[seg], ex);
}

__global__ __launch_bounds__(256) void k_seg_pool(
    const float* __restrict__ h, const float* __restrict__ exn, const float* __restrict__ se,
    const int* __restrict__ batch, const int* __restrict__ ch, float* __restrict__ Z, int n) {
  int node = blockIdx.x * 8 + (threadIdx.x >> 5);
  if (node >= n) return;
  int lane = threadIdx.x & 31;
  int seg = batch[node] * NUM_CH + ch[node];
  float w = exn[node] / (se[seg] + 1e-8f);
  float4 hv = *(const float4*)&h[(size_t)node * H + lane * 4];
  float* zp = &Z[(size_t)seg * H + lane * 4];
  atomicAdd(zp + 0, w * hv.x);
  atomicAdd(zp + 1, w * hv.y);
  atomicAdd(zp + 2, w * hv.z);
  atomicAdd(zp + 3, w * hv.w);
}

// ---------------- tiny 6x6 attention (thread per (graph, q-token, head)) ----------------
__global__ __launch_bounds__(256) void k_cc_attn(
    const float* __restrict__ Q, const float* __restrict__ Km,
    const float* __restrict__ V, float* __restrict__ O, int Bg) {
  int idx = blockIdx.x * 256 + threadIdx.x;
  if (idx >= Bg * NUM_CH * HEADS) return;
  int b = idx / (NUM_CH * HEADS);
  int r = idx % (NUM_CH * HEADS);
  int q = r >> 2, hh = r & 3;
  const float* qp = Q + ((size_t)(b * NUM_CH + q) * H + hh * 32);
  float sc[NUM_CH];
  float mx = -1e30f;
  for (int k = 0; k < NUM_CH; ++k) {
    const float* kp = Km + ((size_t)(b * NUM_CH + k) * H + hh * 32);
    float d = 0.f;
    for (int j = 0; j < 32; ++j) d += qp[j] * kp[j];
    sc[k] = d * 0.17677669529663687f;             // 1/sqrt(32)
    mx = fmaxf(mx, sc[k]);
  }
  float ssum = 0.f;
  for (int k = 0; k < NUM_CH; ++k) { sc[k] = expf(sc[k] - mx); ssum += sc[k]; }
  float inv = 1.f / ssum;
  float* op = O + ((size_t)(b * NUM_CH + q) * H + hh * 32);
  for (int j = 0; j < 32; ++j) {
    float a = 0.f;
    for (int k = 0; k < NUM_CH; ++k)
      a += sc[k] * V[(size_t)(b * NUM_CH + k) * H + hh * 32 + j];
    op[j] = a * inv;
  }
}

__global__ void k_elu_inplace(float* __restrict__ p, size_t n) {
  size_t i = (size_t)blockIdx.x * blockDim.x + threadIdx.x;
  size_t st = (size_t)gridDim.x * blockDim.x;
  for (; i < n; i += st) p[i] = eluf(p[i]);
}

__global__ __launch_bounds__(256) void k_final(
    const float* __restrict__ P1, const float* __restrict__ w2,
    const float* __restrict__ b2, float* __restrict__ out, int Bg) {
  int b = blockIdx.x * 8 + (threadIdx.x >> 5);
  if (b >= Bg) return;
  int lane = threadIdx.x & 31;
  float a = 0.f;
  for (int j = lane; j < RO_HID; j += 32) a += eluf(P1[(size_t)b * RO_HID + j]) * w2[j];
  a = warp_sum(a);
  if (lane == 0) out[b] = a + b2[0];
}

// ---------------- fills ----------------
__global__ void k_fill_f32(float* __restrict__ p, float v, size_t n) {
  size_t i = (size_t)blockIdx.x * blockDim.x + threadIdx.x;
  size_t st = (size_t)gridDim.x * blockDim.x;
  for (; i < n; i += st) p[i] = v;
}
__global__ void k_fill_u32(unsigned* __restrict__ p, unsigned v, size_t n) {
  size_t i = (size_t)blockIdx.x * blockDim.x + threadIdx.x;
  size_t st = (size_t)gridDim.x * blockDim.x;
  for (; i < n; i += st) p[i] = v;
}

// ---------------- host side ----------------
static inline unsigned fenc_host(float x) {
  unsigned u; memcpy(&u, &x, 4);
  return (u & 0x80000000u) ? ~u : (u | 0x80000000u);
}

extern "C" void kernel_launch(void* const* d_in, const int* in_sizes, int n_in,
                              void* d_out, int out_size, void* d_ws, size_t ws_size,
                              hipStream_t stream) {
  (void)in_sizes; (void)n_in; (void)out_size; (void)ws_size;
  // ---- input mapping: top-level dict insertion order; params flattened pytree
  // (sorted dict keys, lists in index order) ----
  const float* x    = (const float*)d_in[0];
  const int*   edge = (const int*)d_in[1];
  const int*   srcI = edge;
  const int*   dstI = edge + E_EDGES;
  const int*   ch   = (const int*)d_in[2];
  const int*   bat  = (const int*)d_in[3];
  auto P = [&](int i) { return (const float*)d_in[i]; };
  // cc[l]: Wk, Wo, Wq, Wv, ff_W1, ff_W2, ff_b1, ff_b2, ln1_b, ln1_g, ln2_b, ln2_g
  struct CCP { const float *Wk,*Wo,*Wq,*Wv,*fW1,*fW2,*fb1,*fb2,*l1b,*l1g,*l2b,*l2g; };
  CCP cc[2];
  for (int l = 0; l < 2; ++l) {
    int o = 4 + 12 * l;
    cc[l] = {P(o),P(o+1),P(o+2),P(o+3),P(o+4),P(o+5),P(o+6),P(o+7),P(o+8),P(o+9),P(o+10),P(o+11)};
  }
  const float *encW = P(28), *encb = P(29), *encbeta = P(30), *encg = P(31);
  struct GATP { const float *W,*adst,*asrc,*b; };
  GATP gat[3];
  for (int l = 0; l < 3; ++l) { int o = 32 + 4 * l; gat[l] = {P(o),P(o+1),P(o+2),P(o+3)}; }
  const float *prW1 = P(44), *prW2 = P(45), *prb1 = P(46), *prb2 = P(47);
  const float *roW1 = P(48), *roW2 = P(49), *rob1 = P(50), *rob2 = P(51);

  // ---- workspace layout (floats) ----
  float* pool = (float*)d_ws;
  size_t off = 0;
  float* h    = pool + off; off += (size_t)N_NODES * H;     // 25.95M
  float* hW   = pool + off; off += (size_t)N_NODES * H;
  float* acc  = pool + off; off += (size_t)N_NODES * H;
  float* as_  = pool + off; off += (size_t)N_NODES * 4;
  float* ad_  = pool + off; off += (size_t)N_NODES * 4;
  unsigned* mx = (unsigned*)(pool + off); off += (size_t)N_NODES * 4;
  float* den  = pool + off; off += (size_t)N_NODES * 4;
  float* exbuf = pool + off; off += (size_t)E_EDGES * 4;
  unsigned* mxseg = (unsigned*)(pool + off); off += NSEG;
  float* seseg = pool + off; off += NSEG;
  float* Z    = pool + off; off += (size_t)NSEG * H;
  // phase-2 reuse of GAT buffers
  float* T   = hW;                        // N x 64 readout hidden
  float* sS  = acc;                       // N readout scores
  float* exn = acc + N_NODES;             // N
  float* Qb  = acc;                       // NSEG x 128 each
  float* Kb  = acc + (size_t)1 * NSEG * H;
  float* Vb  = acc + (size_t)2 * NSEG * H;
  float* Ob  = acc + (size_t)3 * NSEG * H;
  float* O2b = acc + (size_t)4 * NSEG * H;
  float* F1  = acc + (size_t)5 * NSEG * H; // NSEG x 512
  float* P1  = hW;                         // B x 256

  auto gemm = [&](const float* A, const float* W, const float* bias, float* C,
                  int M, int K, int Nc) {
    dim3 g((M + 63) / 64, Nc / 64);
    k_gemm_f16<<<g, dim3(128), 0, stream>>>(A, W, bias, C, M, K, Nc);
  };
  auto fillf = [&](float* p, float v, size_t n) {
    int blocks = (int)(((n + 255) / 256 < 4096) ? (n + 255) / 256 : 4096);
    k_fill_f32<<<blocks, 256, 0, stream>>>(p, v, n);
  };
  auto fillu = [&](unsigned* p, unsigned v, size_t n) {
    int blocks = (int)(((n + 255) / 256 < 4096) ? (n + 255) / 256 : 4096);
    k_fill_u32<<<blocks, 256, 0, stream>>>(p, v, n);
  };

  const int nodeWaves = (N_NODES + 7) / 8;
  const int edgeThr   = (E_EDGES + 255) / 256;
  const int edgeWaves = (E_EDGES + 7) / 8;

  // ---- 1. encoder: h = ELU(LN(x @ encW + encb)) ----
  gemm(x, encW, encb, h, N_NODES, F_IN, H);
  k_ln<<<nodeWaves, 256, 0, stream>>>(h, nullptr, encg, encbeta, N_NODES, 1);

  // ---- 2. three GAT layers ----
  for (int l = 0; l < 3; ++l) {
    gemm(h, gat[l].W, nullptr, hW, N_NODES, H, H);
    k_gat_scores<<<nodeWaves, 256, 0, stream>>>(hW, gat[l].asrc, gat[l].adst, as_, ad_, N_NODES);
    fillu(mx, fenc_host(NEGC), (size_t)N_NODES * 4);
    fillf(den, 0.f, (size_t)N_NODES * 4);
    fillf(acc, 0.f, (size_t)N_NODES * H);
    k_edge_max<<<edgeThr, 256, 0, stream>>>(srcI, dstI, ch, as_, ad_, mx, E_EDGES);
    k_edge_exp<<<edgeThr, 256, 0, stream>>>(srcI, dstI, ch, as_, ad_, mx, exbuf, den, E_EDGES);
    k_edge_agg<<<edgeWaves, 256, 0, stream>>>(srcI, dstI, ch, hW, exbuf, den, acc, E_EDGES);
    k_gat_post<<<4096, 256, 0, stream>>>(h, acc, gat[l].b, (size_t)N_NODES * H);
  }

  // ---- 3. attention pooling into (graph, channel) tokens ----
  gemm(h, roW1, rob1, T, N_NODES, H, H / 2);
  k_ro_score<<<nodeWaves, 256, 0, stream>>>(T, roW2, rob2, sS, N_NODES);
  fillu(mxseg, fenc_host(-__builtin_inff()), NSEG);
  k_seg_max<<<(N_NODES + 255) / 256, 256, 0, stream>>>(sS, bat, ch, mxseg, N_NODES);
  fillf(seseg, 0.f, NSEG);
  k_seg_expsum<<<(N_NODES + 255) / 256, 256, 0, stream>>>(sS, bat, ch, mxseg, exn, seseg, N_NODES);
  fillf(Z, 0.f, (size_t)NSEG * H);
  k_seg_pool<<<nodeWaves, 256, 0, stream>>>(h, exn, seseg, bat, ch, Z, N_NODES);

  // ---- 4. cross-channel transformer (2 layers over 6 tokens/graph) ----
  for (int l = 0; l < 2; ++l) {
    gemm(Z, cc[l].Wq, nullptr, Qb, NSEG, H, H);
    gemm(Z, cc[l].Wk, nullptr, Kb, NSEG, H, H);
    gemm(Z, cc[l].Wv, nullptr, Vb, NSEG, H, H);
    k_cc_attn<<<(B_GRAPHS * NUM_CH * HEADS + 255) / 256, 256, 0, stream>>>(Qb, Kb, Vb, Ob, B_GRAPHS);
    gemm(Ob, cc[l].Wo, nullptr, O2b, NSEG, H, H);
    k_ln<<<(NSEG + 7) / 8, 256, 0, stream>>>(Z, O2b, cc[l].l1g, cc[l].l1b, NSEG, 0);
    gemm(Z, cc[l].fW1, cc[l].fb1, F1, NSEG, H, FF_DIM);
    k_elu_inplace<<<4096, 256, 0, stream>>>(F1, (size_t)NSEG * FF_DIM);
    gemm(F1, cc[l].fW2, cc[l].fb2, O2b, NSEG, FF_DIM, H);
    k_ln<<<(NSEG + 7) / 8, 256, 0, stream>>>(Z, O2b, cc[l].l2g, cc[l].l2b, NSEG, 0);
  }

  // ---- 5. prediction head: out = ELU(flat @ prW1 + b1) @ prW2 + b2 ----
  gemm(Z, prW1, prb1, P1, B_GRAPHS, NUM_CH * H, RO_HID);   // Z memory layout == flat
  k_final<<<(B_GRAPHS + 7) / 8, 256, 0, stream>>>(P1, prW2, prb2, (float*)d_out, B_GRAPHS);
}
